// Qwen3MoeDecoderLayer_55405078118964
// MI455X (gfx1250) — compile-verified, matrix operand order checked by reading, not run on hardware
//
#include <hip/hip_runtime.h>

#define T_TOK 1024
#define HDIM  2048
#define NHEAD 32
#define NKVH  4
#define DHEAD 128
#define NEXP  64
#define KTOP  8
#define IDIM  768
#define QKVN  ((NHEAD + 2 * NKVH) * DHEAD) /* 5120 */

typedef __attribute__((ext_vector_type(16))) __bf16        v16bf;
typedef __attribute__((ext_vector_type(8)))  float         v8f;
typedef __attribute__((ext_vector_type(8)))  unsigned int  v8u;

union FragCvt { v8u u; v16bf b; };

#ifndef __has_builtin
#define __has_builtin(x) 0
#endif
#if defined(__HIP_DEVICE_COMPILE__) &&                                         \
    __has_builtin(__builtin_amdgcn_tensor_load_to_lds) &&                      \
    __has_builtin(__builtin_amdgcn_s_wait_tensorcnt)
#define USE_TDM 1
#else
#define USE_TDM 0
#endif

__device__ __forceinline__ unsigned short f2bf(float f) {
  unsigned int u = __float_as_uint(f);
  unsigned int r = u + 0x7FFFu + ((u >> 16) & 1u); // round-to-nearest-even
  return (unsigned short)(r >> 16);
}

__device__ __forceinline__ v8f v8f_zero() {
  v8f z = {0.f, 0.f, 0.f, 0.f, 0.f, 0.f, 0.f, 0.f};
  return z;
}

// Load a 16x32 bf16 WMMA fragment (A-layout; B mirrored, rows=N) from an LDS
// tile stored row-major with strideU16 halfwords/row. strideU16, kOff even.
__device__ __forceinline__ v16bf ldsfrag(const unsigned short* base, int row0,
                                         int strideU16, int kOff) {
  int lane = threadIdx.x & 31;
  int r = row0 + (lane & 15);
  int kb = (lane < 16) ? 0 : 8;
  const unsigned int* p = (const unsigned int*)base;
  int ro = r * strideU16 + kOff + kb;
  FragCvt f;
#pragma unroll
  for (int v = 0; v < 8; ++v) {
    int k = (v < 4) ? (2 * v) : (16 + 2 * (v - 4));
    f.u[v] = p[(ro + k) >> 1];
  }
  return f.b;
}

__device__ __forceinline__ v8f wmma_bf16(v16bf a, v16bf b, v8f c) {
  return __builtin_amdgcn_wmma_f32_16x16x32_bf16(false, a, false, b,
                                                 (short)0, c, false, false);
}

#if USE_TDM
typedef __attribute__((ext_vector_type(4))) unsigned int tdm_v4u;
typedef __attribute__((ext_vector_type(8))) int          tdm_v8i;
typedef __attribute__((ext_vector_type(4))) int          tdm_v4i;

// 2D bf16 tile DMA: global (tileH rows x tileW elems, rowStride elems apart)
// -> LDS at ldsAddr; LDS rows = (1<<(padCode+1)) DWORDs + (padAmt+1) DWORDs.
// Rows >= tensorRows are zero-filled (OOB reads return 0).
__device__ __forceinline__ void tdm_load_tile_bf16(
    unsigned ldsAddr, const unsigned short* gsrc, unsigned tileW,
    unsigned tileH, unsigned rowStride, unsigned tensorRows, unsigned padCode,
    unsigned padAmt) {
  unsigned long long ga = (unsigned long long)(size_t)gsrc;
  tdm_v4u g0;
  g0[0] = 1u;                                            // count=1 user D#
  g0[1] = ldsAddr;                                       // LDS byte address
  g0[2] = (unsigned)ga;                                  // global addr lo
  g0[3] = (unsigned)((ga >> 32) & 0x01FFFFFFull) | (2u << 30); // hi | type=2
  tdm_v8i g1;
  g1[0] = (int)((1u << 16) | (1u << 20) | (padCode << 22) | (padAmt << 25));
  g1[1] = (int)((tileW & 0xFFFFu) << 16);                // tensor_dim0 lo
  g1[2] = (int)((tileW >> 16) | ((tensorRows & 0xFFFFu) << 16));
  g1[3] = (int)((tensorRows >> 16) | ((tileW & 0xFFFFu) << 16)); // tile_dim0
  g1[4] = (int)(tileH & 0xFFFFu);                        // tile_dim1
  g1[5] = (int)rowStride;                                // dim0_stride lo32
  g1[6] = 0;
  g1[7] = 0;
  tdm_v4i gz = {0, 0, 0, 0};
#if __clang_major__ >= 23
  tdm_v8i gz8 = {0, 0, 0, 0, 0, 0, 0, 0};
  __builtin_amdgcn_tensor_load_to_lds(g0, g1, gz, gz, gz8, 0);
#else
  __builtin_amdgcn_tensor_load_to_lds(g0, g1, gz, gz, 0);
#endif
}
#endif // USE_TDM

// ---------------------------------------------------------------------------
// RMSNorm row kernel: out_bf16 = x * rsqrt(mean(x^2)+eps) * w  (vectorized)
// ---------------------------------------------------------------------------
__global__ __launch_bounds__(256) void rmsnorm_k(const float* __restrict__ x,
                                                 const float* __restrict__ w,
                                                 unsigned short* __restrict__ out,
                                                 int ncols) {
  int row = blockIdx.x;
  const float* xr = x + (size_t)row * ncols;
  __shared__ float red[256];
  float ss = 0.f;
  for (int c = threadIdx.x * 4; c < ncols; c += 1024) {
    const float4 xv = *(const float4*)&xr[c];
    ss += xv.x * xv.x + xv.y * xv.y + xv.z * xv.z + xv.w * xv.w;
  }
  red[threadIdx.x] = ss;
  __syncthreads();
  for (int off = 128; off > 0; off >>= 1) {
    if (threadIdx.x < off) red[threadIdx.x] += red[threadIdx.x + off];
    __syncthreads();
  }
  float inv = rsqrtf(red[0] / (float)ncols + 1e-6f);
  for (int c = threadIdx.x * 4; c < ncols; c += 1024) {
    const float4 xv = *(const float4*)&xr[c];
    const float4 wv = *(const float4*)&w[c];
    uint2 o;
    o.x = (unsigned)f2bf(xv.x * inv * wv.x) |
          ((unsigned)f2bf(xv.y * inv * wv.y) << 16);
    o.y = (unsigned)f2bf(xv.z * inv * wv.z) |
          ((unsigned)f2bf(xv.w * inv * wv.w) << 16);
    *(uint2*)&out[(size_t)row * ncols + c] = o;
  }
}

// ---------------------------------------------------------------------------
// Generic bf16 WMMA GEMM: C(f32)[M,N] = A_bf16[M,K] * B_f32->bf16[K,N] (+Res)
// 64x64 tile per block, 4 waves, K-step 64, TDM-staged A, b128 B loads.
// ---------------------------------------------------------------------------
template <int EPI> // 0 = store f32, 1 = store f32 + residual add
__global__ __launch_bounds__(128) void gemm_bf16_k(
    const unsigned short* __restrict__ A, const float* __restrict__ B,
    float* __restrict__ C, const float* __restrict__ Res, int M, int N, int Kd) {
  __shared__ unsigned short As[64][68]; // [m][k]
  __shared__ unsigned short Bs[64][68]; // [n][k] (B^T)
  int tid = threadIdx.x;
  int lane = tid & 31, wid = tid >> 5;
  int wr = wid >> 1, wc = wid & 1;
  int m0 = blockIdx.y * 64, n0 = blockIdx.x * 64;

  v8f acc[2][2];
#pragma unroll
  for (int i = 0; i < 2; ++i)
#pragma unroll
    for (int j = 0; j < 2; ++j) acc[i][j] = v8f_zero();

  for (int kb = 0; kb < Kd; kb += 64) {
    __syncthreads();
#if USE_TDM
    if (wid == 0)
      tdm_load_tile_bf16((unsigned)(size_t)&As[0][0],
                         A + (size_t)m0 * Kd + kb, 64u, 64u, (unsigned)Kd,
                         64u, 4u /* 32-DW rows */, 1u /* +2 DW pad */);
#else
    for (int idx = tid; idx < 64 * 16; idx += 128) {
      int r = idx >> 4, k4 = (idx & 15) << 2;
      uint2 av = *(const uint2*)&A[(size_t)(m0 + r) * Kd + kb + k4];
      *(uint2*)&As[r][k4] = av;
    }
#endif
    for (int idx = tid; idx < 64 * 16; idx += 128) {
      int k = idx >> 4, n4 = (idx & 15) << 2;
      const float4 bv = *(const float4*)&B[(size_t)(kb + k) * N + n0 + n4];
      Bs[n4 + 0][k] = f2bf(bv.x);
      Bs[n4 + 1][k] = f2bf(bv.y);
      Bs[n4 + 2][k] = f2bf(bv.z);
      Bs[n4 + 3][k] = f2bf(bv.w);
      if (kb + 64 < Kd)
        __builtin_prefetch(&B[(size_t)(kb + 64 + k) * N + n0 + n4], 0, 0);
    }
#if USE_TDM
    if (wid == 0) __builtin_amdgcn_s_wait_tensorcnt(0);
#endif
    __syncthreads();
#pragma unroll
    for (int kk = 0; kk < 64; kk += 32) {
      v16bf a0 = ldsfrag(&As[0][0], wr * 32, 68, kk);
      v16bf a1 = ldsfrag(&As[0][0], wr * 32 + 16, 68, kk);
      v16bf b0 = ldsfrag(&Bs[0][0], wc * 32, 68, kk);
      v16bf b1 = ldsfrag(&Bs[0][0], wc * 32 + 16, 68, kk);
      acc[0][0] = wmma_bf16(a0, b0, acc[0][0]);
      acc[0][1] = wmma_bf16(a0, b1, acc[0][1]);
      acc[1][0] = wmma_bf16(a1, b0, acc[1][0]);
      acc[1][1] = wmma_bf16(a1, b1, acc[1][1]);
    }
  }

  int half8 = (lane >= 16) ? 8 : 0;
  int cn = lane & 15;
#pragma unroll
  for (int i = 0; i < 2; ++i)
#pragma unroll
    for (int j = 0; j < 2; ++j)
#pragma unroll
      for (int v = 0; v < 8; ++v) {
        int row = m0 + wr * 32 + i * 16 + v + half8;
        int col = n0 + wc * 32 + j * 16 + cn;
        float val = acc[i][j][v];
        if constexpr (EPI == 1) val += Res[(size_t)row * N + col];
        C[(size_t)row * N + col] = val;
      }
}

// ---------------------------------------------------------------------------
// Per-(token, head) Q/K rmsnorm + RoPE; V passthrough. blockDim.x = 128 (=D).
// ---------------------------------------------------------------------------
__global__ __launch_bounds__(128) void qkrope_k(
    const float* __restrict__ qkv, const float* __restrict__ qw,
    const float* __restrict__ kw, const int* __restrict__ pos,
    unsigned short* __restrict__ Qb, unsigned short* __restrict__ Kb,
    unsigned short* __restrict__ Vb) {
  int t = blockIdx.x, hy = blockIdx.y, d = threadIdx.x;
  const int KO = NHEAD * DHEAD, VO = (NHEAD + NKVH) * DHEAD;
  if (hy >= NHEAD + NKVH) { // V head: just convert
    int vh = hy - NHEAD - NKVH;
    float v = qkv[(size_t)t * QKVN + VO + vh * DHEAD + d];
    Vb[((size_t)t * NKVH + vh) * DHEAD + d] = f2bf(v);
    return;
  }
  __shared__ float buf[DHEAD];
  __shared__ float red[4];
  float val;
  if (hy < NHEAD)
    val = qkv[(size_t)t * QKVN + hy * DHEAD + d];
  else
    val = qkv[(size_t)t * QKVN + KO + (hy - NHEAD) * DHEAD + d];
  float ss = val * val;
#pragma unroll
  for (int off = 16; off > 0; off >>= 1) ss += __shfl_xor(ss, off, 32);
  if ((d & 31) == 0) red[d >> 5] = ss;
  __syncthreads();
  float tot = red[0] + red[1] + red[2] + red[3];
  float inv = rsqrtf(tot / (float)DHEAD + 1e-6f);
  const float* w = (hy < NHEAD) ? qw : kw;
  buf[d] = val * inv * w[d];
  __syncthreads();
  int i = d & 63;
  float p = (float)pos[t];
  float freq = __powf(1.0e6f, -((float)(2 * i) / (float)DHEAD));
  float s, c;
  __sincosf(p * freq, &s, &c);
  float x1 = buf[i], x2 = buf[i + 64];
  float o = (d < 64) ? (x1 * c - x2 * s) : (x2 * c + x1 * s);
  if (hy < NHEAD)
    Qb[((size_t)t * NHEAD + hy) * DHEAD + d] = f2bf(o);
  else
    Kb[((size_t)t * NKVH + (hy - NHEAD)) * DHEAD + d] = f2bf(o);
}

// ---------------------------------------------------------------------------
// Flash attention, causal, GQA. Block = (head, 64-row q tile), 4 waves.
// QK^T and PV use bf16 WMMA; Q/K tiles staged by TDM; online softmax in f32.
// ---------------------------------------------------------------------------
__global__ __launch_bounds__(128) void attn_k(const unsigned short* __restrict__ Q,
                                              const unsigned short* __restrict__ Kt,
                                              const unsigned short* __restrict__ Vt,
                                              unsigned short* __restrict__ O) {
  int h = blockIdx.x, qt = blockIdx.y;
  int kvh = h >> 3; // NHEAD / NKVH = 8
  int tid = threadIdx.x, lane = tid & 31, wid = tid >> 5;
  int half8 = (lane >= 16) ? 8 : 0;
  int cn = lane & 15;

  __shared__ unsigned short Qs[64][DHEAD + 4];
  __shared__ unsigned short Ks[32][DHEAD + 4];
  __shared__ unsigned short Vst[DHEAD][36]; // transposed [d][key]
  __shared__ unsigned short Ps[4][16][36];  // per-wave P tile

#if USE_TDM
  if (wid == 0)
    tdm_load_tile_bf16((unsigned)(size_t)&Qs[0][0],
                       Q + ((size_t)(qt * 64) * NHEAD + h) * DHEAD, DHEAD, 64u,
                       NHEAD * DHEAD, 64u, 5u /* 64-DW rows */, 1u);
#else
  for (int idx = tid; idx < 64 * 32; idx += 128) {
    int m = idx >> 5, d4 = (idx & 31) << 2;
    uint2 qv = *(const uint2*)&Q[((size_t)(qt * 64 + m) * NHEAD + h) * DHEAD + d4];
    *(uint2*)&Qs[m][d4] = qv;
  }
#endif

  v8f o[8];
  float m8[8], l8[8];
#pragma unroll
  for (int n = 0; n < 8; ++n) o[n] = v8f_zero();
#pragma unroll
  for (int v = 0; v < 8; ++v) { m8[v] = -1e30f; l8[v] = 0.f; }

  const float scale = 0.08838834764831845f; // D^-0.5
  int ktEnd = (qt + 1) * 2;                 // causal: keys <= qt*64+63
  for (int kt = 0; kt < ktEnd; ++kt) {
    __syncthreads();
#if USE_TDM
    if (wid == 0)
      tdm_load_tile_bf16((unsigned)(size_t)&Ks[0][0],
                         Kt + ((size_t)(kt * 32) * NKVH + kvh) * DHEAD, DHEAD,
                         32u, NKVH * DHEAD, 32u, 5u, 1u);
#else
    for (int idx = tid; idx < 32 * 32; idx += 128) {
      int r = idx >> 5, d4 = (idx & 31) << 2;
      uint2 kv =
          *(const uint2*)&Kt[((size_t)(kt * 32 + r) * NKVH + kvh) * DHEAD + d4];
      *(uint2*)&Ks[r][d4] = kv;
    }
#endif
    for (int idx = tid; idx < 32 * 32; idx += 128) {
      int r = idx >> 5, d4 = (idx & 31) << 2;
      uint2 vv =
          *(const uint2*)&Vt[((size_t)(kt * 32 + r) * NKVH + kvh) * DHEAD + d4];
      Vst[d4 + 0][r] = (unsigned short)(vv.x & 0xFFFFu);
      Vst[d4 + 1][r] = (unsigned short)(vv.x >> 16);
      Vst[d4 + 2][r] = (unsigned short)(vv.y & 0xFFFFu);
      Vst[d4 + 3][r] = (unsigned short)(vv.y >> 16);
    }
#if USE_TDM
    if (wid == 0) __builtin_amdgcn_s_wait_tensorcnt(0);
#endif
    __syncthreads();

    v8f c0 = v8f_zero(), c1 = v8f_zero();
#pragma unroll
    for (int kk = 0; kk < 4; ++kk) {
      v16bf a = ldsfrag(&Qs[0][0], wid * 16, DHEAD + 4, kk * 32);
      v16bf b0 = ldsfrag(&Ks[0][0], 0, DHEAD + 4, kk * 32);
      v16bf b1 = ldsfrag(&Ks[0][0], 16, DHEAD + 4, kk * 32);
      c0 = wmma_bf16(a, b0, c0);
      c1 = wmma_bf16(a, b1, c1);
    }

    int qrowBase = qt * 64 + wid * 16 + half8;
    int key0 = kt * 32 + cn, key1 = key0 + 16;
#pragma unroll
    for (int v = 0; v < 8; ++v) {
      int qrow = qrowBase + v;
      float s0 = c0[v] * scale;
      float s1 = c1[v] * scale;
      if (key0 > qrow) s0 = -1e30f;
      if (key1 > qrow) s1 = -1e30f;
      float x = fmaxf(s0, s1);
#pragma unroll
      for (int off = 1; off <= 8; off <<= 1) x = fmaxf(x, __shfl_xor(x, off, 32));
      float mn = fmaxf(m8[v], x);
      float r = __expf(m8[v] - mn);
      float p0 = __expf(s0 - mn);
      float p1 = __expf(s1 - mn);
      float rs = p0 + p1;
#pragma unroll
      for (int off = 1; off <= 8; off <<= 1) rs += __shfl_xor(rs, off, 32);
      l8[v] = l8[v] * r + rs;
      m8[v] = mn;
#pragma unroll
      for (int n = 0; n < 8; ++n) o[n] *= r;
      Ps[wid][v + half8][cn] = f2bf(p0);
      Ps[wid][v + half8][cn + 16] = f2bf(p1);
    }
    __syncthreads();

    v16bf pa = ldsfrag(&Ps[wid][0][0], 0, 36, 0);
#pragma unroll
    for (int n = 0; n < 8; ++n) {
      v16bf b = ldsfrag(&Vst[0][0], n * 16, 36, 0);
      o[n] = wmma_bf16(pa, b, o[n]);
    }
  }

#pragma unroll
  for (int n = 0; n < 8; ++n)
#pragma unroll
    for (int v = 0; v < 8; ++v) {
      int qrow = qt * 64 + wid * 16 + v + half8;
      float val = o[n][v] / l8[v];
      O[(size_t)qrow * (NHEAD * DHEAD) + h * DHEAD + n * 16 + cn] = f2bf(val);
    }
}

// ---------------------------------------------------------------------------
// Router top-k + per-expert token lists
// ---------------------------------------------------------------------------
__global__ void zero_k(int* cnt) {
  if (threadIdx.x < NEXP) cnt[threadIdx.x] = 0;
}

__global__ __launch_bounds__(256) void topk_k(const float* __restrict__ logits,
                                              int* cnt, int* tokList,
                                              float* gateVal, int* tokE,
                                              int* tokPos) {
  int t = blockIdx.x * 256 + threadIdx.x;
  if (t >= T_TOK) return;
  float l[NEXP];
  float mx = -1e30f;
#pragma unroll 8
  for (int e = 0; e < NEXP; ++e) {
    l[e] = logits[(size_t)t * NEXP + e];
    mx = fmaxf(mx, l[e]);
  }
  int ids[KTOP];
  float vals[KTOP];
  float sum = 0.f;
  for (int j = 0; j < KTOP; ++j) {
    int be = 0;
    float bv = -1e30f;
#pragma unroll 8
    for (int e = 0; e < NEXP; ++e)
      if (l[e] > bv) { bv = l[e]; be = e; }
    float p = __expf(bv - mx);
    ids[j] = be;
    vals[j] = p;
    sum += p;
    l[be] = -1e31f;
  }
  for (int j = 0; j < KTOP; ++j) {
    int e = ids[j];
    int p = atomicAdd(&cnt[e], 1);
    tokList[e * T_TOK + p] = t;
    gateVal[e * T_TOK + p] = vals[j] / sum;
    tokE[t * KTOP + j] = e;
    tokPos[t * KTOP + j] = p;
  }
}

__global__ void prefix_k(const int* cnt, int* offs) {
  if (threadIdx.x == 0) {
    int a = 0;
    for (int e = 0; e < NEXP; ++e) { offs[e] = a; a += cnt[e]; }
    offs[NEXP] = a;
  }
}

// ---------------------------------------------------------------------------
// MoE phase A: act[slot, i] = silu(Xe @ wg) * (Xe @ wu), token-gathered GEMM
// ---------------------------------------------------------------------------
__global__ __launch_bounds__(128) void moeA_k(
    const unsigned short* __restrict__ Xn, const float* __restrict__ Wg,
    const float* __restrict__ Wu, const int* __restrict__ tokList,
    const int* __restrict__ cnt, const int* __restrict__ offs,
    unsigned short* __restrict__ Act) {
  int e = blockIdx.z;
  int c = cnt[e];
  int rt = blockIdx.x;
  if (rt * 64 >= c) return;
  int ib = blockIdx.y * 64;

  __shared__ unsigned short As[64][68];
  __shared__ unsigned short Bg[64][68];
  __shared__ unsigned short Bu[64][68];
  __shared__ int rowTok[64];

  int tid = threadIdx.x, lane = tid & 31, wid = tid >> 5;
  int wr = wid >> 1, wc = wid & 1;
  if (tid < 64) {
    int s = rt * 64 + tid;
    rowTok[tid] = (s < c) ? tokList[e * T_TOK + s] : -1;
  }
  const float* wg = Wg + (size_t)e * HDIM * IDIM;
  const float* wu = Wu + (size_t)e * HDIM * IDIM;

  v8f aG[2][2], aU[2][2];
#pragma unroll
  for (int i = 0; i < 2; ++i)
#pragma unroll
    for (int j = 0; j < 2; ++j) { aG[i][j] = v8f_zero(); aU[i][j] = v8f_zero(); }

  __syncthreads();
  for (int kb = 0; kb < HDIM; kb += 64) {
    __syncthreads();
    for (int idx = tid; idx < 64 * 16; idx += 128) {
      int r = idx >> 4, k4 = (idx & 15) << 2;
      int tok = rowTok[r];
      uint2 av = make_uint2(0u, 0u);
      if (tok >= 0) av = *(const uint2*)&Xn[(size_t)tok * HDIM + kb + k4];
      *(uint2*)&As[r][k4] = av;
    }
    for (int idx = tid; idx < 64 * 16; idx += 128) {
      int k = idx >> 4, n4 = (idx & 15) << 2;
      size_t go = (size_t)(kb + k) * IDIM + ib + n4;
      const float4 gv = *(const float4*)&wg[go];
      const float4 uv = *(const float4*)&wu[go];
      Bg[n4 + 0][k] = f2bf(gv.x);
      Bg[n4 + 1][k] = f2bf(gv.y);
      Bg[n4 + 2][k] = f2bf(gv.z);
      Bg[n4 + 3][k] = f2bf(gv.w);
      Bu[n4 + 0][k] = f2bf(uv.x);
      Bu[n4 + 1][k] = f2bf(uv.y);
      Bu[n4 + 2][k] = f2bf(uv.z);
      Bu[n4 + 3][k] = f2bf(uv.w);
      if (kb + 64 < HDIM) {
        size_t gn = (size_t)(kb + 64 + k) * IDIM + ib + n4;
        __builtin_prefetch(&wg[gn], 0, 0);
        __builtin_prefetch(&wu[gn], 0, 0);
      }
    }
    __syncthreads();
#pragma unroll
    for (int kk = 0; kk < 64; kk += 32) {
      v16bf a0 = ldsfrag(&As[0][0], wr * 32, 68, kk);
      v16bf a1 = ldsfrag(&As[0][0], wr * 32 + 16, 68, kk);
      v16bf g0 = ldsfrag(&Bg[0][0], wc * 32, 68, kk);
      v16bf g1 = ldsfrag(&Bg[0][0], wc * 32 + 16, 68, kk);
      v16bf u0 = ldsfrag(&Bu[0][0], wc * 32, 68, kk);
      v16bf u1 = ldsfrag(&Bu[0][0], wc * 32 + 16, 68, kk);
      aG[0][0] = wmma_bf16(a0, g0, aG[0][0]);
      aG[0][1] = wmma_bf16(a0, g1, aG[0][1]);
      aG[1][0] = wmma_bf16(a1, g0, aG[1][0]);
      aG[1][1] = wmma_bf16(a1, g1, aG[1][1]);
      aU[0][0] = wmma_bf16(a0, u0, aU[0][0]);
      aU[0][1] = wmma_bf16(a0, u1, aU[0][1]);
      aU[1][0] = wmma_bf16(a1, u0, aU[1][0]);
      aU[1][1] = wmma_bf16(a1, u1, aU[1][1]);
    }
  }

  int half8 = (lane >= 16) ? 8 : 0;
  int cn = lane & 15;
  int base = offs[e];
#pragma unroll
  for (int i = 0; i < 2; ++i)
#pragma unroll
    for (int j = 0; j < 2; ++j)
#pragma unroll
      for (int v = 0; v < 8; ++v) {
        int s = rt * 64 + wr * 32 + i * 16 + v + half8;
        if (s < c) {
          float g = aG[i][j][v], u = aU[i][j][v];
          float act = (g / (1.f + __expf(-g))) * u;
          Act[(size_t)(base + s) * IDIM + ib + wc * 32 + j * 16 + cn] = f2bf(act);
        }
      }
}

// ---------------------------------------------------------------------------
// MoE phase B: yslot = (act @ wd) * gate, per-slot (no atomics). A tile rows
// are contiguous slots -> TDM-staged with OOB zero fill via tensor_dim1.
// ---------------------------------------------------------------------------
__global__ __launch_bounds__(128) void moeB_k(
    const unsigned short* __restrict__ Act, const float* __restrict__ Wd,
    const float* __restrict__ gateVal, const int* __restrict__ cnt,
    const int* __restrict__ offs, float* __restrict__ Yslot) {
  int e = blockIdx.z;
  int c = cnt[e];
  int rt = blockIdx.x;
  if (rt * 64 >= c) return;
  int nb = blockIdx.y * 64;

  __shared__ unsigned short As[64][68];
  __shared__ unsigned short Bs[64][68];
  __shared__ float gvs[64];

  int tid = threadIdx.x, lane = tid & 31, wid = tid >> 5;
  int wr = wid >> 1, wc = wid & 1;
  int base = offs[e];
  if (tid < 64) {
    int s = rt * 64 + tid;
    gvs[tid] = (s < c) ? gateVal[e * T_TOK + s] : 0.f;
  }
  const float* wd = Wd + (size_t)e * IDIM * HDIM;

  v8f acc[2][2];
#pragma unroll
  for (int i = 0; i < 2; ++i)
#pragma unroll
    for (int j = 0; j < 2; ++j) acc[i][j] = v8f_zero();

  __syncthreads();
  for (int kb = 0; kb < IDIM; kb += 64) {
    __syncthreads();
#if USE_TDM
    if (wid == 0)
      tdm_load_tile_bf16((unsigned)(size_t)&As[0][0],
                         Act + (size_t)(base + rt * 64) * IDIM + kb, 64u, 64u,
                         IDIM, (unsigned)(c - rt * 64), 4u, 1u);
#else
    for (int idx = tid; idx < 64 * 16; idx += 128) {
      int r = idx >> 4, k4 = (idx & 15) << 2;
      int s = rt * 64 + r;
      uint2 av = make_uint2(0u, 0u);
      if (s < c) av = *(const uint2*)&Act[(size_t)(base + s) * IDIM + kb + k4];
      *(uint2*)&As[r][k4] = av;
    }
#endif
    for (int idx = tid; idx < 64 * 16; idx += 128) {
      int k = idx >> 4, n4 = (idx & 15) << 2;
      const float4 bv = *(const float4*)&wd[(size_t)(kb + k) * HDIM + nb + n4];
      Bs[n4 + 0][k] = f2bf(bv.x);
      Bs[n4 + 1][k] = f2bf(bv.y);
      Bs[n4 + 2][k] = f2bf(bv.z);
      Bs[n4 + 3][k] = f2bf(bv.w);
      if (kb + 64 < IDIM)
        __builtin_prefetch(&wd[(size_t)(kb + 64 + k) * HDIM + nb + n4], 0, 0);
    }
#if USE_TDM
    if (wid == 0) __builtin_amdgcn_s_wait_tensorcnt(0);
#endif
    __syncthreads();
#pragma unroll
    for (int kk = 0; kk < 64; kk += 32) {
      v16bf a0 = ldsfrag(&As[0][0], wr * 32, 68, kk);
      v16bf a1 = ldsfrag(&As[0][0], wr * 32 + 16, 68, kk);
      v16bf b0 = ldsfrag(&Bs[0][0], wc * 32, 68, kk);
      v16bf b1 = ldsfrag(&Bs[0][0], wc * 32 + 16, 68, kk);
      acc[0][0] = wmma_bf16(a0, b0, acc[0][0]);
      acc[0][1] = wmma_bf16(a0, b1, acc[0][1]);
      acc[1][0] = wmma_bf16(a1, b0, acc[1][0]);
      acc[1][1] = wmma_bf16(a1, b1, acc[1][1]);
    }
  }

  int half8 = (lane >= 16) ? 8 : 0;
  int cn = lane & 15;
#pragma unroll
  for (int i = 0; i < 2; ++i)
#pragma unroll
    for (int j = 0; j < 2; ++j)
#pragma unroll
      for (int v = 0; v < 8; ++v) {
        int lr = wr * 32 + i * 16 + v + half8;
        int s = rt * 64 + lr;
        if (s < c) {
          float val = acc[i][j][v] * gvs[lr];
          Yslot[(size_t)(base + s) * HDIM + nb + wc * 32 + j * 16 + cn] = val;
        }
      }
}

// ---------------------------------------------------------------------------
// Final per-token reduction: out = hidden + sum_{j<8} yslot[slot_j]
// ---------------------------------------------------------------------------
__global__ __launch_bounds__(256) void reduce_k(
    const float* __restrict__ hidden, const float* __restrict__ Yslot,
    const int* __restrict__ tokE, const int* __restrict__ tokPos,
    const int* __restrict__ offs, float* __restrict__ out) {
  int t = blockIdx.x;
  __shared__ int slots[KTOP];
  if (threadIdx.x < KTOP) {
    int e = tokE[t * KTOP + threadIdx.x];
    slots[threadIdx.x] = offs[e] + tokPos[t * KTOP + threadIdx.x];
  }
  __syncthreads();
  for (int c = threadIdx.x * 4; c < HDIM; c += 1024) {
    float4 a = *(const float4*)&hidden[(size_t)t * HDIM + c];
#pragma unroll
    for (int j = 0; j < KTOP; ++j) {
      const float4 y = *(const float4*)&Yslot[(size_t)slots[j] * HDIM + c];
      a.x += y.x; a.y += y.y; a.z += y.z; a.w += y.w;
    }
    *(float4*)&out[(size_t)t * HDIM + c] = a;
  }
}

// ---------------------------------------------------------------------------
extern "C" void kernel_launch(void* const* d_in, const int* in_sizes, int n_in,
                              void* d_out, int out_size, void* d_ws,
                              size_t ws_size, hipStream_t stream) {
  const float* hs   = (const float*)d_in[0];
  const int*   pos  = (const int*)d_in[1];
  const float* wqkv = (const float*)d_in[2];
  const float* qnw  = (const float*)d_in[3];
  const float* knw  = (const float*)d_in[4];
  const float* wo   = (const float*)d_in[5];
  const float* ln1  = (const float*)d_in[6];
  const float* ln2  = (const float*)d_in[7];
  const float* gw   = (const float*)d_in[8];
  const float* wg   = (const float*)d_in[9];
  const float* wu   = (const float*)d_in[10];
  const float* wd   = (const float*)d_in[11];
  float* out = (float*)d_out;

  char* ws = (char*)d_ws;
  size_t off = 0;
  auto alloc = [&](size_t bytes) -> void* {
    void* p = ws + off;
    off += (bytes + 255) & ~(size_t)255;
    return p;
  };

  unsigned short* xn     = (unsigned short*)alloc((size_t)T_TOK * HDIM * 2);
  float*          qkv    = (float*)alloc((size_t)T_TOK * QKVN * 4);
  unsigned short* qb     = (unsigned short*)alloc((size_t)T_TOK * NHEAD * DHEAD * 2);
  unsigned short* kb     = (unsigned short*)alloc((size_t)T_TOK * NKVH * DHEAD * 2);
  unsigned short* vb     = (unsigned short*)alloc((size_t)T_TOK * NKVH * DHEAD * 2);
  unsigned short* attnb  = (unsigned short*)alloc((size_t)T_TOK * NHEAD * DHEAD * 2);
  float*          hidden = (float*)alloc((size_t)T_TOK * HDIM * 4);
  unsigned short* x2n    = (unsigned short*)alloc((size_t)T_TOK * HDIM * 2);
  float*          logits = (float*)alloc((size_t)T_TOK * NEXP * 4);
  int*            cnt    = (int*)alloc(NEXP * 4);
  int*            offsB  = (int*)alloc((NEXP + 1) * 4);
  int*            tokL   = (int*)alloc((size_t)NEXP * T_TOK * 4);
  float*          gateV  = (float*)alloc((size_t)NEXP * T_TOK * 4);
  int*            tokE   = (int*)alloc((size_t)T_TOK * KTOP * 4);
  int*            tokP   = (int*)alloc((size_t)T_TOK * KTOP * 4);
  unsigned short* act    = (unsigned short*)alloc((size_t)T_TOK * KTOP * IDIM * 2);
  float*          yslot  = (float*)alloc((size_t)T_TOK * KTOP * HDIM * 4);

  // 1) ln1 -> bf16
  rmsnorm_k<<<T_TOK, 256, 0, stream>>>(hs, ln1, xn, HDIM);
  // 2) qkv = xn @ wqkv
  gemm_bf16_k<0><<<dim3(QKVN / 64, T_TOK / 64), 128, 0, stream>>>(
      xn, wqkv, qkv, nullptr, T_TOK, QKVN, HDIM);
  // 3) q/k rmsnorm + rope, v convert
  qkrope_k<<<dim3(T_TOK, NHEAD + 2 * NKVH), 128, 0, stream>>>(
      qkv, qnw, knw, pos, qb, kb, vb);
  // 4) flash attention
  attn_k<<<dim3(NHEAD, T_TOK / 64), 128, 0, stream>>>(qb, kb, vb, attnb);
  // 5) hidden = residual + attn @ wo
  gemm_bf16_k<1><<<dim3(HDIM / 64, T_TOK / 64), 128, 0, stream>>>(
      attnb, wo, hidden, hs, T_TOK, HDIM, NHEAD * DHEAD);
  // 6) ln2 -> bf16
  rmsnorm_k<<<T_TOK, 256, 0, stream>>>(hidden, ln2, x2n, HDIM);
  // 7) router logits = x2n @ gate_w
  gemm_bf16_k<0><<<dim3(NEXP / 64, T_TOK / 64), 128, 0, stream>>>(
      x2n, gw, logits, nullptr, T_TOK, NEXP, HDIM);
  // 8) routing
  zero_k<<<1, 64, 0, stream>>>(cnt);
  topk_k<<<T_TOK / 256, 256, 0, stream>>>(logits, cnt, tokL, gateV, tokE, tokP);
  prefix_k<<<1, 1, 0, stream>>>(cnt, offsB);
  // 9) MoE gate/up + silu
  moeA_k<<<dim3(T_TOK / 64, IDIM / 64, NEXP), 128, 0, stream>>>(
      x2n, wg, wu, tokL, cnt, offsB, act);
  // 10) MoE down + gate scale
  moeB_k<<<dim3(T_TOK / 64, HDIM / 64, NEXP), 128, 0, stream>>>(
      act, wd, gateV, cnt, offsB, yslot);
  // 11) out = hidden + sum of expert slots
  reduce_k<<<T_TOK, 256, 0, stream>>>(hidden, yslot, tokE, tokP, offsB, out);

  (void)in_sizes; (void)n_in; (void)out_size; (void)ws_size;
}
